// MoEFFN_HKV_22703197127137
// MI455X (gfx1250) — compile-verified, matrix operand order checked by reading, not run on hardware
//
#include <hip/hip_runtime.h>
#include <math.h>

// ---------------------------------------------------------------------------
// MoE FFN with hierarchical router for MI455X (gfx1250, wave32, WMMA).
// bf16 WMMA 16x16x32 path; weights pre-transposed so both A and B fragments
// load with two ds/global b128 loads per lane in the documented 16-bit layout.
// X tiles staged into LDS with GLOBAL_LOAD_ASYNC_TO_LDS_B128 (ASYNCcnt) when
// the toolchain exposes the builtins; VGPR-bounce fallback otherwise.
// ---------------------------------------------------------------------------

typedef __bf16 bf16;
typedef __attribute__((ext_vector_type(16))) __bf16 v16bf;
typedef __attribute__((ext_vector_type(8)))  float  v8f;
typedef __attribute__((ext_vector_type(4)))  unsigned int u32x4;
typedef __attribute__((ext_vector_type(4)))  int v4i;

#if __has_builtin(__builtin_amdgcn_global_load_async_to_lds_b128) && \
    __has_builtin(__builtin_amdgcn_s_wait_asynccnt)
#define HAVE_ASYNC_LDS 1
#else
#define HAVE_ASYNC_LDS 0
#endif

#define NB    4      // buckets
#define EPB   4      // experts per bucket
#define NE    16     // total experts
#define TOPK  2
#define CC    512
#define HH    1024
#define NT    2048   // B*T tokens
#define MT    32     // token rows per tile
#define HCHUNK 128   // hidden chunk staged in LDS
#define LDX   (CC + 8)
#define LDH   (HCHUNK + 8)

union FragU { v16bf v; u32x4 q[2]; };

#if HAVE_ASYNC_LDS
typedef __attribute__((address_space(1))) v4i gv4i;   // global b128 payload
typedef __attribute__((address_space(3))) v4i lv4i;   // LDS b128 payload
__device__ __forceinline__ void async_copy16(const bf16* g, bf16* l) {
  __builtin_amdgcn_global_load_async_to_lds_b128(
      (gv4i*)(v4i*)(void*)const_cast<bf16*>(g),
      (lv4i*)(v4i*)(void*)l,
      0, 0);
}
#endif

// 16-bit A/B fragment (16x32) from a row-major [16][ld] tile.
// lane<16 : M=lane,    K = {0..7, 16..23}
// lane>=16: M=lane-16, K = {8..15, 24..31}
__device__ __forceinline__ v16bf load_frag(const bf16* base, int ld, int lane) {
  const int m  = lane & 15;
  const int kb = (lane & 16) ? 8 : 0;
  const bf16* p = base + m * ld + kb;
  FragU f;
  f.q[0] = *(const u32x4*)(p);
  f.q[1] = *(const u32x4*)(p + 16);
  return f.v;
}

__device__ __forceinline__ v8f wmma_bf16(v16bf a, v16bf b, v8f c) {
  return __builtin_amdgcn_wmma_f32_16x16x32_bf16(false, a, false, b, (short)0, c, false, false);
}

// --------------------------- conversion kernels ----------------------------

__global__ void k_conv(const float* __restrict__ src, bf16* __restrict__ dst, int n) {
  int i = blockIdx.x * blockDim.x + threadIdx.x;
  if (i < n) dst[i] = (bf16)src[i];
}

// src is [rows][cols] row-major, dst is [cols][rows] (transposed), bf16.
// blockIdx.y selects the expert (matrix) slice.
__global__ void k_conv_t(const float* __restrict__ src, bf16* __restrict__ dst,
                         int rows, int cols) {
  long base = (long)blockIdx.y * rows * cols;
  int i = blockIdx.x * blockDim.x + threadIdx.x;
  if (i < rows * cols) {
    int r = i / cols, c = i % cols;
    dst[base + (long)c * rows + r] = (bf16)src[base + i];
  }
}

// ------------------------------- router ------------------------------------

__global__ void k_keynorm(const float* __restrict__ ek, float* __restrict__ kinv) {
  int wv = threadIdx.x >> 5, lane = threadIdx.x & 31;
  if (wv < NE) {
    float s = 0.f;
    for (int c = lane; c < CC; c += 32) { float v = ek[wv * CC + c]; s += v * v; }
    for (int o = 16; o; o >>= 1) s += __shfl_xor(s, o, 32);
    if (lane == 0) kinv[wv] = 1.0f / fmaxf(sqrtf(s), 1e-12f);
  }
}

__global__ void k_init(int* cnt, int* cursor) {
  int i = threadIdx.x;
  if (i < NE) { cnt[i] = 0; cursor[i] = 0; }
}

__global__ void k_router(const float* __restrict__ x, const int* __restrict__ op_id,
                         const float* __restrict__ ek, const float* __restrict__ kinv,
                         int* __restrict__ gid, float* __restrict__ wts, int* cnt) {
  int wv = threadIdx.x >> 5, lane = threadIdx.x & 31;
  int t = blockIdx.x * (blockDim.x >> 5) + wv;
  if (t >= NT) return;
  int b = op_id[t]; b = b < 0 ? 0 : (b > NB - 1 ? NB - 1 : b);
  const float* xt = x + (long)t * CC;
  const float* kp = ek + (long)b * EPB * CC;
  float xs = 0.f, d0 = 0.f, d1 = 0.f, d2 = 0.f, d3 = 0.f;
  for (int c = lane; c < CC; c += 32) {
    float xv = xt[c];
    xs += xv * xv;
    d0 += xv * kp[c];
    d1 += xv * kp[CC + c];
    d2 += xv * kp[2 * CC + c];
    d3 += xv * kp[3 * CC + c];
  }
  for (int o = 16; o; o >>= 1) {
    xs += __shfl_xor(xs, o, 32);
    d0 += __shfl_xor(d0, o, 32);
    d1 += __shfl_xor(d1, o, 32);
    d2 += __shfl_xor(d2, o, 32);
    d3 += __shfl_xor(d3, o, 32);
  }
  if (lane == 0) {
    float invx = 1.0f / fmaxf(sqrtf(xs), 1e-12f);
    float s[4] = { d0 * kinv[b * EPB + 0] * invx, d1 * kinv[b * EPB + 1] * invx,
                   d2 * kinv[b * EPB + 2] * invx, d3 * kinv[b * EPB + 3] * invx };
    float m = fmaxf(fmaxf(s[0], s[1]), fmaxf(s[2], s[3]));
    float p[4], sum = 0.f;
    for (int e = 0; e < 4; e++) { p[e] = __expf(s[e] - m); sum += p[e]; }
    for (int e = 0; e < 4; e++) p[e] /= sum;
    int i0 = 0;
    for (int e = 1; e < 4; e++) if (p[e] > p[i0]) i0 = e;
    int i1 = (i0 == 0) ? 1 : 0;
    for (int e = 0; e < 4; e++) if (e != i0 && p[e] > p[i1]) i1 = e;
    float denom = p[i0] + p[i1] + 1e-9f;
    int g0 = b * EPB + i0, g1 = b * EPB + i1;
    gid[t * 2]     = g0; wts[t * 2]     = p[i0] / denom;
    gid[t * 2 + 1] = g1; wts[t * 2 + 1] = p[i1] / denom;
    atomicAdd(&cnt[g0], 1); atomicAdd(&cnt[g1], 1);
  }
}

__global__ void k_scan(const int* __restrict__ cnt, int* __restrict__ off) {
  if (threadIdx.x == 0) {
    int s = 0;
    for (int e = 0; e < NE; e++) { off[e] = s; s += cnt[e]; }
  }
}

__global__ void k_scatter(const int* __restrict__ gid, const int* __restrict__ off,
                          int* cursor, int* __restrict__ tok_of, int* __restrict__ pos_tk) {
  int t = blockIdx.x * blockDim.x + threadIdx.x;
  if (t >= NT) return;
  for (int k = 0; k < TOPK; k++) {
    int e = gid[t * 2 + k];
    int slot = atomicAdd(&cursor[e], 1);
    int pos = off[e] + slot;
    tok_of[pos] = t;
    pos_tk[t * 2 + k] = pos;
  }
}

// --------------------- fused FFN (dense shared path) -----------------------
// 32-row tile per block, 8 waves. Stage X(32x512) in LDS; per 128-wide H
// chunk: WMMA GEMM1 -> relu+bias -> bf16 in LDS -> WMMA GEMM2 accumulate.

__global__ __launch_bounds__(256) void k_dense(const bf16* __restrict__ xb,
                                               const bf16* __restrict__ w1t, const float* __restrict__ b1,
                                               const bf16* __restrict__ w2t, const float* __restrict__ b2,
                                               float* __restrict__ dout) {
  __shared__ bf16 Xs[MT * LDX];
  __shared__ bf16 Hs[MT * LDH];
  const int wv = threadIdx.x >> 5, lane = threadIdx.x & 31;
  const int row0 = blockIdx.x * MT;

  const int chunks = MT * CC / 8;  // 16B chunks
#if HAVE_ASYNC_LDS
  for (int i = threadIdx.x; i < chunks; i += blockDim.x) {
    int r = i >> 6;
    int c = (i & 63) << 3;
    async_copy16(&xb[(long)(row0 + r) * CC + c], &Xs[r * LDX + c]);
  }
  __builtin_amdgcn_s_wait_asynccnt(0);
  __syncthreads();
#else
  for (int i = threadIdx.x; i < chunks; i += blockDim.x) {
    int r = i >> 6;
    int c = (i & 63) << 3;
    *(u32x4*)&Xs[r * LDX + c] = *(const u32x4*)&xb[(long)(row0 + r) * CC + c];
  }
  __syncthreads();
#endif

  v8f acc2[2][4] = {};
  for (int hc = 0; hc < HH; hc += HCHUNK) {
    v8f acc1[2] = {};
    const bf16* w1base = w1t + (long)(hc + wv * 16) * CC;
#pragma unroll 4
    for (int kc = 0; kc < CC; kc += 32) {
      __builtin_prefetch(w1base + kc + 256, 0, 1);
      v16bf a0 = load_frag(&Xs[0 * LDX + kc], LDX, lane);
      v16bf a1 = load_frag(&Xs[16 * LDX + kc], LDX, lane);
      v16bf bb = load_frag(w1base + kc, CC, lane);
      acc1[0] = wmma_bf16(a0, bb, acc1[0]);
      acc1[1] = wmma_bf16(a1, bb, acc1[1]);
    }
    {
      int n = lane & 15;
      float bias = b1[hc + wv * 16 + n];
      int rbase = (lane & 16) ? 8 : 0;
      int klocal = wv * 16 + n;
#pragma unroll
      for (int mi = 0; mi < 2; mi++)
#pragma unroll
        for (int v = 0; v < 8; v++) {
          float h = fmaxf(acc1[mi][v] + bias, 0.0f);
          Hs[(mi * 16 + rbase + v) * LDH + klocal] = (bf16)h;
        }
    }
    __syncthreads();
    const bf16* w2base = w2t + (long)(wv * 64) * HH + hc;
#pragma unroll
    for (int kh = 0; kh < HCHUNK; kh += 32) {
      v16bf a0 = load_frag(&Hs[0 * LDH + kh], LDH, lane);
      v16bf a1 = load_frag(&Hs[16 * LDH + kh], LDH, lane);
#pragma unroll
      for (int nt = 0; nt < 4; nt++) {
        v16bf bb = load_frag(w2base + (long)nt * 16 * HH + kh, HH, lane);
        acc2[0][nt] = wmma_bf16(a0, bb, acc2[0][nt]);
        acc2[1][nt] = wmma_bf16(a1, bb, acc2[1][nt]);
      }
    }
    __syncthreads();
  }
  {
    int n = lane & 15;
    int rbase = (lane & 16) ? 8 : 0;
#pragma unroll
    for (int mi = 0; mi < 2; mi++)
#pragma unroll
      for (int nt = 0; nt < 4; nt++) {
        int col = wv * 64 + nt * 16 + n;
        float bias = b2[col];
#pragma unroll
        for (int v = 0; v < 8; v++) {
          int r = row0 + mi * 16 + rbase + v;
          dout[(long)r * CC + col] = acc2[mi][nt][v] + bias;
        }
      }
  }
}

// ----------------- fused FFN (grouped expert path, gathered) ---------------

__global__ __launch_bounds__(256) void k_expert(const bf16* __restrict__ xb,
                                                const bf16* __restrict__ ew1t, const float* __restrict__ eb1,
                                                const bf16* __restrict__ ew2t, const float* __restrict__ eb2,
                                                const int* __restrict__ cnt, const int* __restrict__ off,
                                                const int* __restrict__ tok_of,
                                                float* __restrict__ oa) {
  const int e = blockIdx.y;
  const int cnt_e = cnt[e];
  const int tile = blockIdx.x;
  if (tile * MT >= cnt_e) return;
  const int base = off[e];

  __shared__ bf16 Xs[MT * LDX];
  __shared__ bf16 Hs[MT * LDH];
  const int wv = threadIdx.x >> 5, lane = threadIdx.x & 31;

  const int chunks = MT * CC / 8;
#if HAVE_ASYNC_LDS
  for (int i = threadIdx.x; i < chunks; i += blockDim.x) {
    int r = i >> 6;
    int c = (i & 63) << 3;
    int ridx = tile * MT + r;
    if (ridx < cnt_e) {
      int tok = tok_of[base + ridx];
      async_copy16(&xb[(long)tok * CC + c], &Xs[r * LDX + c]);
    } else {
      u32x4 z = {0u, 0u, 0u, 0u};
      *(u32x4*)&Xs[r * LDX + c] = z;
    }
  }
  __builtin_amdgcn_s_wait_asynccnt(0);
  __syncthreads();
#else
  for (int i = threadIdx.x; i < chunks; i += blockDim.x) {
    int r = i >> 6;
    int c = (i & 63) << 3;
    int ridx = tile * MT + r;
    u32x4 val = {0u, 0u, 0u, 0u};
    if (ridx < cnt_e) {
      int tok = tok_of[base + ridx];
      val = *(const u32x4*)&xb[(long)tok * CC + c];
    }
    *(u32x4*)&Xs[r * LDX + c] = val;
  }
  __syncthreads();
#endif

  const bf16* w1t = ew1t + (long)e * HH * CC;  // [H][C]
  const bf16* w2t = ew2t + (long)e * CC * HH;  // [C][H]
  const float* b1 = eb1 + e * HH;
  const float* b2 = eb2 + e * CC;

  v8f acc2[2][4] = {};
  for (int hc = 0; hc < HH; hc += HCHUNK) {
    v8f acc1[2] = {};
    const bf16* w1base = w1t + (long)(hc + wv * 16) * CC;
#pragma unroll 4
    for (int kc = 0; kc < CC; kc += 32) {
      __builtin_prefetch(w1base + kc + 256, 0, 1);
      v16bf a0 = load_frag(&Xs[0 * LDX + kc], LDX, lane);
      v16bf a1 = load_frag(&Xs[16 * LDX + kc], LDX, lane);
      v16bf bb = load_frag(w1base + kc, CC, lane);
      acc1[0] = wmma_bf16(a0, bb, acc1[0]);
      acc1[1] = wmma_bf16(a1, bb, acc1[1]);
    }
    {
      int n = lane & 15;
      float bias = b1[hc + wv * 16 + n];
      int rbase = (lane & 16) ? 8 : 0;
      int klocal = wv * 16 + n;
#pragma unroll
      for (int mi = 0; mi < 2; mi++)
#pragma unroll
        for (int v = 0; v < 8; v++) {
          float h = fmaxf(acc1[mi][v] + bias, 0.0f);
          Hs[(mi * 16 + rbase + v) * LDH + klocal] = (bf16)h;
        }
    }
    __syncthreads();
    const bf16* w2base = w2t + (long)(wv * 64) * HH + hc;
#pragma unroll
    for (int kh = 0; kh < HCHUNK; kh += 32) {
      v16bf a0 = load_frag(&Hs[0 * LDH + kh], LDH, lane);
      v16bf a1 = load_frag(&Hs[16 * LDH + kh], LDH, lane);
#pragma unroll
      for (int nt = 0; nt < 4; nt++) {
        v16bf bb = load_frag(w2base + (long)nt * 16 * HH + kh, HH, lane);
        acc2[0][nt] = wmma_bf16(a0, bb, acc2[0][nt]);
        acc2[1][nt] = wmma_bf16(a1, bb, acc2[1][nt]);
      }
    }
    __syncthreads();
  }
  {
    int n = lane & 15;
    int rbase = (lane & 16) ? 8 : 0;
#pragma unroll
    for (int mi = 0; mi < 2; mi++)
#pragma unroll
      for (int nt = 0; nt < 4; nt++) {
        int col = wv * 64 + nt * 16 + n;
        float bias = b2[col];
#pragma unroll
        for (int v = 0; v < 8; v++) {
          int ridx = tile * MT + mi * 16 + rbase + v;
          if (ridx < cnt_e)
            oa[(long)(base + ridx) * CC + col] = acc2[mi][nt][v] + bias;
        }
      }
  }
}

// ------------------------------- combine -----------------------------------

__global__ void k_combine(const float* __restrict__ dense, const float* __restrict__ oa,
                          const int* __restrict__ pos_tk, const float* __restrict__ wts,
                          const float* __restrict__ gate, float* __restrict__ out) {
  int i = blockIdx.x * blockDim.x + threadIdx.x;
  if (i >= NT * CC) return;
  int t = i >> 9;           // / CC
  int c = i & (CC - 1);
  float g = 1.0f / (1.0f + __expf(-gate[0]));
  int p0 = pos_tk[t * 2], p1 = pos_tk[t * 2 + 1];
  float m = wts[t * 2] * oa[(long)p0 * CC + c] + wts[t * 2 + 1] * oa[(long)p1 * CC + c];
  out[i] = dense[i] + g * m;
}

// ------------------------------- launch ------------------------------------

extern "C" void kernel_launch(void* const* d_in, const int* in_sizes, int n_in,
                              void* d_out, int out_size, void* d_ws, size_t ws_size,
                              hipStream_t stream) {
  (void)in_sizes; (void)n_in; (void)out_size; (void)ws_size;
  const float* x    = (const float*)d_in[0];
  const int*   op   = (const int*)d_in[1];
  const float* ek   = (const float*)d_in[2];
  const float* sW1  = (const float*)d_in[3];
  const float* sb1  = (const float*)d_in[4];
  const float* sW2  = (const float*)d_in[5];
  const float* sb2  = (const float*)d_in[6];
  const float* eW1  = (const float*)d_in[7];
  const float* eb1  = (const float*)d_in[8];
  const float* eW2  = (const float*)d_in[9];
  const float* eb2  = (const float*)d_in[10];
  const float* gate = (const float*)d_in[11];
  float* out = (float*)d_out;

  char* ws = (char*)d_ws;
  size_t o = 0;
  auto alloc = [&](size_t bytes) -> void* {
    o = (o + 255) & ~(size_t)255;
    void* p = ws + o;
    o += bytes;
    return p;
  };

  bf16*  xb     = (bf16*)alloc((size_t)NT * CC * 2);
  bf16*  sW1t   = (bf16*)alloc((size_t)CC * HH * 2);           // [H][C]
  bf16*  sW2t   = (bf16*)alloc((size_t)CC * HH * 2);           // [C][H]
  bf16*  eW1t   = (bf16*)alloc((size_t)NE * CC * HH * 2);      // [E][H][C]
  bf16*  eW2t   = (bf16*)alloc((size_t)NE * CC * HH * 2);      // [E][C][H]
  float* kinv   = (float*)alloc(NE * 4);
  int*   gid    = (int*)alloc(NT * 2 * 4);
  float* wts    = (float*)alloc(NT * 2 * 4);
  int*   cnt    = (int*)alloc(NE * 4);
  int*   offp   = (int*)alloc(NE * 4);
  int*   cursor = (int*)alloc(NE * 4);
  int*   tok_of = (int*)alloc(NT * 2 * 4);
  int*   pos_tk = (int*)alloc(NT * 2 * 4);
  float* oa     = (float*)alloc((size_t)NT * 2 * CC * 4);
  float* dres   = (float*)alloc((size_t)NT * CC * 4);

  // precision / layout prep
  k_conv  <<<(NT * CC + 255) / 256, 256, 0, stream>>>(x, xb, NT * CC);
  k_conv_t<<<dim3((CC * HH + 255) / 256, 1),  256, 0, stream>>>(sW1, sW1t, CC, HH);
  k_conv_t<<<dim3((CC * HH + 255) / 256, 1),  256, 0, stream>>>(sW2, sW2t, HH, CC);
  k_conv_t<<<dim3((CC * HH + 255) / 256, NE), 256, 0, stream>>>(eW1, eW1t, CC, HH);
  k_conv_t<<<dim3((CC * HH + 255) / 256, NE), 256, 0, stream>>>(eW2, eW2t, HH, CC);

  // routing
  k_keynorm<<<1, 512, 0, stream>>>(ek, kinv);
  k_init   <<<1, 32, 0, stream>>>(cnt, cursor);
  k_router <<<NT / 8, 256, 0, stream>>>(x, op, ek, kinv, gid, wts, cnt);
  k_scan   <<<1, 32, 0, stream>>>(cnt, offp);
  k_scatter<<<NT / 256, 256, 0, stream>>>(gid, offp, cursor, tok_of, pos_tk);

  // FFNs
  k_dense <<<NT / MT, 256, 0, stream>>>(xb, sW1t, sb1, sW2t, sb2, dres);
  k_expert<<<dim3(NT * 2 / MT, NE), 256, 0, stream>>>(xb, eW1t, eb1, eW2t, eb2,
                                                      cnt, offp, tok_of, oa);

  // gated combine
  k_combine<<<(NT * CC + 255) / 256, 256, 0, stream>>>(dres, oa, pos_tk, wts, gate, out);
}
